// Head_395136991264
// MI455X (gfx1250) — matrix operand order repaired and checked by reading, not compiled, not run on hardware
//
#include <hip/hip_runtime.h>

#define BB 4
#define TT 4096
#define CC 384
#define HH 64

typedef __attribute__((ext_vector_type(16))) __bf16 v16bf;
typedef __attribute__((ext_vector_type(8)))  float  v8f;

union FragBF { v16bf v; unsigned int u[8]; };
union Pack2 { unsigned int u; __bf16 h[2]; };

// Raw v_exp_f32: no libm denormal-range fixup.  Safe here: |s| is tiny in the
// main loop, and v_exp_f32(-inf) == +0 for masked tail entries.
__device__ __forceinline__ float fast_exp2(float x) {
  return __builtin_amdgcn_exp2f(x);
}

// ---------------------------------------------------------------------------
// Kernel 1: projections.  K = x@Wk^T + bk (bf16 [B][T][H]),
// Qs = (x@Wv^T + bv) * 384^-0.5 * log2(e)  (bf16 [B][T][H]; Q == V per the
// reference bug; scale and log2e folded in so attention uses exp2 directly),
// Vt = (x@Wv^T + bv) transposed (bf16 [B][H][T]) so the PV WMMA B-operand
// (K-dim = time) has contiguous packed bf16 pairs.
// ---------------------------------------------------------------------------
__global__ __launch_bounds__(256) void proj_kernel(
    const float* __restrict__ x, const float* __restrict__ Wk,
    const float* __restrict__ bk, const float* __restrict__ Wv,
    const float* __restrict__ bv, __bf16* __restrict__ Kbf,
    __bf16* __restrict__ Qs, __bf16* __restrict__ Vt) {
  __shared__ float xs[4][CC];
  const int row0 = blockIdx.x * 4;  // 4 rows of (B*T) per block
  for (int i = threadIdx.x; i < 4 * CC; i += 256)
    xs[i / CC][i % CC] = x[(size_t)row0 * CC + i];
  __syncthreads();

  const int h = threadIdx.x & 63;
  const int r = threadIdx.x >> 6;  // 0..3
  const int row = row0 + r;
  const int b = row / TT;
  const int t = row % TT;
  // 384^-0.5 * log2(e): scores come out in log2 domain -> P = exp2(s).
  const float qscale = 0.05103103630798287f * 1.4426950408889634f;

  const float4* wk4 = (const float4*)(Wk + (size_t)h * CC);
  const float4* wv4 = (const float4*)(Wv + (size_t)h * CC);
  const float4* xr4 = (const float4*)(xs[r]);
  float ak = 0.f, av = 0.f;
#pragma unroll 4
  for (int c = 0; c < CC / 4; ++c) {
    float4 xv = xr4[c];
    float4 k4 = wk4[c];
    float4 v4 = wv4[c];
    ak += xv.x * k4.x + xv.y * k4.y + xv.z * k4.z + xv.w * k4.w;
    av += xv.x * v4.x + xv.y * v4.y + xv.z * v4.z + xv.w * v4.w;
  }
  ak += bk[h];
  av += bv[h];
  const size_t idx = (size_t)row * HH + h;
  Kbf[idx] = (__bf16)ak;
  Qs[idx] = (__bf16)(av * qscale);
  Vt[(size_t)b * HH * TT + (size_t)h * TT + t] = (__bf16)av;
}

// ---------------------------------------------------------------------------
// One 32-key block for 32 query rows (2 M-tiles).  K/V fragments are loaded
// once and shared by both tiles.  No running max: scores are provably tiny
// (|s| < ~5), so P = exp2(s) cannot overflow and softmax is exact.
// Row sums accumulate via an extra WMMA against a ones matrix.
//
// Key interleave: S-fragment sfr covers keys kb + 2n + sfr (column n), so a
// lane's two P values (cols ln of both fragments) are K-adjacent keys
// (kb+2ln, kb+2ln+1) -> single packed b32 LDS store; the A-layout reload
// pairs stay contiguous because LDS storage is natural key order.
// MASKED=false: hot path (no clamps / cndmask).  MASKED=true: causal tail.
// ---------------------------------------------------------------------------
template <bool MASKED>
__device__ __forceinline__ void attn_block(
    int kb, int t0, int half, int ln, const __bf16* __restrict__ Kb,
    const __bf16* __restrict__ Vtb, const FragBF (&qf)[2][2], v8f (&o)[2][4],
    v8f (&lacc)[2], v16bf ones, unsigned int (*pl)[16][17]) {
  // ---- S = Q K^T : 2 key N-tiles x 2 head-dim K-chunks x 2 M-tiles ----
  v8f s[2][2];
#pragma unroll
  for (int t = 0; t < 2; ++t)
#pragma unroll
    for (int f = 0; f < 2; ++f) s[t][f] = (v8f){};

#pragma unroll
  for (int sfr = 0; sfr < 2; ++sfr) {
    int key = kb + 2 * ln + sfr;  // B-layout: lane = N; interleaved keys
    if (MASKED && key > TT - 1) key = TT - 1;
    const unsigned int* krow = (const unsigned int*)(Kb + (size_t)key * HH);
#pragma unroll
    for (int ch = 0; ch < 2; ++ch) {
      FragBF kf;
#pragma unroll
      for (int v = 0; v < 8; ++v) kf.u[v] = krow[ch * 16 + ((v < 4) ? (half * 4 + v) : (8 + half * 4 + (v - 4)))];
#pragma unroll
      for (int t = 0; t < 2; ++t)
        s[t][sfr] = __builtin_amdgcn_wmma_f32_16x16x32_bf16(
            false, qf[t][ch].v, false, kf.v, (short)0, s[t][sfr], false,
            false);
    }
  }

  // ---- P = exp2(S) (causal mask in tail); packed bf16 pair -> LDS ----
#pragma unroll
  for (int t = 0; t < 2; ++t) {
#pragma unroll
    for (int r = 0; r < 8; ++r) {
      if (MASKED) {
        const int row = t0 + t * 16 + r + half * 8;
        if (kb + 2 * ln > row) s[t][0][r] = -INFINITY;
        if (kb + 2 * ln + 1 > row) s[t][1][r] = -INFINITY;
      }
      Pack2 pk;
      pk.h[0] = (__bf16)fast_exp2(s[t][0][r]);  // key kb+2ln   (even K-idx)
      pk.h[1] = (__bf16)fast_exp2(s[t][1][r]);  // key kb+2ln+1 (odd K-idx)
      pl[t][r + half * 8][ln] = pk.u;
    }
  }

  // ---- reload P as A-layout fragments (same-wave LDS is in-order) ----
  FragBF pf[2];
#pragma unroll
  for (int t = 0; t < 2; ++t) {
    const unsigned int* prow = pl[t][ln];
#pragma unroll
    for (int v = 0; v < 8; ++v) {
      int pos = (v < 4) ? (half * 4 + v) : (8 + half * 4 + (v - 4));
      pf[t].u[v] = prow[pos];
    }
  }

  // ---- row sums via WMMA against ones (lands in C-row layout) ----
#pragma unroll
  for (int t = 0; t < 2; ++t)
    lacc[t] = __builtin_amdgcn_wmma_f32_16x16x32_bf16(
        false, pf[t].v, false, ones, (short)0, lacc[t], false, false);

  // ---- O += P V : 4 head-dim N-tiles, B from transposed V ----
#pragma unroll
  for (int ht = 0; ht < 4; ++ht) {
    FragBF vf;
    const __bf16* vcol = Vtb + (size_t)(ht * 16 + ln) * TT;
#pragma unroll
    for (int v = 0; v < 8; ++v) {
      int tb = kb + ((v < 4) ? (half * 8 + v * 2) : (16 + half * 8 + (v - 4) * 2));
      if (MASKED && tb > TT - 2) tb = TT - 2;  // clamped lanes have P == 0
      vf.u[v] = *(const unsigned int*)(vcol + tb);
    }
#pragma unroll
    for (int t = 0; t < 2; ++t)
      o[t][ht] = __builtin_amdgcn_wmma_f32_16x16x32_bf16(
          false, pf[t].v, false, vf.v, (short)0, o[t][ht], false, false);
  }
}

// ---------------------------------------------------------------------------
// Kernel 2: flash attention.  One wave = 32 query rows (2 WMMA M-tiles),
// 18 WMMAs per 32-key block, unmasked main loop + short causal tail.
// Loop bounds go through readfirstlane: uniform SALU loop control, EXEC
// stays all-ones around the WMMAs as the ISA requires.
// ---------------------------------------------------------------------------
__global__ __launch_bounds__(256) void attn_kernel(
    const __bf16* __restrict__ Kbf, const __bf16* __restrict__ Qs,
    const __bf16* __restrict__ Vt, float* __restrict__ out) {
  __shared__ unsigned int Plds[8][2][16][17];  // per-wave packed P staging

  const int lane = threadIdx.x & 31;
  const int wave = threadIdx.x >> 5;
  const int b = blockIdx.y;
  const int t0 = blockIdx.x * 256 + wave * 32;
  const int t0s = __builtin_amdgcn_readfirstlane(t0);  // wave-uniform copy
  const int half = lane >> 4;  // 0/1
  const int ln = lane & 15;

  const __bf16* Kb = Kbf + (size_t)b * TT * HH;
  const __bf16* Qb = Qs + (size_t)b * TT * HH;
  const __bf16* Vtb = Vt + (size_t)b * HH * TT;
  unsigned int(*pl)[16][17] = Plds[wave];

  // Ones B-fragment for row-sum WMMAs (bf16 1.0 = 0x3F80).
  FragBF ones;
#pragma unroll
  for (int v = 0; v < 8; ++v) ones.u[v] = 0x3F803F80u;

  // Q fragments: A-layout 16x32 bf16, 2 M-tiles x 2 head-dim chunks.
  FragBF qf[2][2];
#pragma unroll
  for (int t = 0; t < 2; ++t) {
    const unsigned int* qrow =
        (const unsigned int*)(Qb + (size_t)(t0 + t * 16 + ln) * HH);
#pragma unroll
    for (int ch = 0; ch < 2; ++ch)
#pragma unroll
      for (int v = 0; v < 8; ++v) {
        int k0 = (v < 4) ? (half * 8 + v * 2) : (16 + half * 8 + (v - 4) * 2);
        qf[t][ch].u[v] = qrow[ch * 16 + (k0 >> 1)];
      }
  }

  v8f o[2][4];
  v8f lacc[2];
#pragma unroll
  for (int t = 0; t < 2; ++t) {
    lacc[t] = (v8f){};
#pragma unroll
    for (int i = 0; i < 4; ++i) o[t][i] = (v8f){};
  }

  const int kbmax = ((t0s + 31) >> 5) << 5;
  int kb = 0;
  // Unmasked main loop: every key in [kb, kb+31] is <= t0 (valid & visible).
  for (; kb + 31 <= t0s; kb += 32) {
    __builtin_prefetch(Kb + (size_t)(kb + 32 + ln) * HH, 0, 3);
    attn_block<false>(kb, t0, half, ln, Kb, Vtb, qf, o, lacc, ones.v, pl);
  }
  // Masked tail (the causal diagonal blocks).
  for (; kb <= kbmax; kb += 32)
    attn_block<true>(kb, t0, half, ln, Kb, Vtb, qf, o, lacc, ones.v, pl);

  // ---- normalize and write fp32 output ----
#pragma unroll
  for (int t = 0; t < 2; ++t)
#pragma unroll
    for (int r = 0; r < 8; ++r) {
      const float inv = 1.0f / lacc[t][r];
      const int trow = t0 + t * 16 + r + half * 8;
#pragma unroll
      for (int ht = 0; ht < 4; ++ht)
        out[((size_t)b * TT + trow) * HH + ht * 16 + ln] = o[t][ht][r] * inv;
    }
}

extern "C" void kernel_launch(void* const* d_in, const int* in_sizes, int n_in,
                              void* d_out, int out_size, void* d_ws,
                              size_t ws_size, hipStream_t stream) {
  const float* x  = (const float*)d_in[0];
  const float* Wk = (const float*)d_in[1];
  const float* bk = (const float*)d_in[2];
  // d_in[3], d_in[4] (Wq, bq) are dead: the reference uses the value
  // projection for the query.
  const float* Wv = (const float*)d_in[5];
  const float* bv = (const float*)d_in[6];
  float* out = (float*)d_out;

  char* ws = (char*)d_ws;
  const size_t nkv = (size_t)BB * TT * HH;  // elements per projected tensor
  __bf16* Kbf = (__bf16*)(ws);
  __bf16* Qs  = (__bf16*)(ws + nkv * 2);
  __bf16* Vt  = (__bf16*)(ws + nkv * 4);

  proj_kernel<<<BB * TT / 4, 256, 0, stream>>>(x, Wk, bk, Wv, bv, Kbf, Qs,
                                               Vt);
  attn_kernel<<<dim3(TT / 256, BB), 256, 0, stream>>>(Kbf, Qs, Vt, out);
}
